// SelfAttention_74620761801184
// MI455X (gfx1250) — matrix operand order repaired and checked by reading, not compiled, not run on hardware
//
#include <hip/hip_runtime.h>
#include <hip/hip_bf16.h>
#include <cstdint>
#include <cstddef>

// Problem constants from the reference: B=4, S=2048, H=1024, R=1, HEAD_SIZE=64
#define BB 4
#define SS 2048
#define HH 1024

typedef __attribute__((ext_vector_type(16))) __bf16 v16bf;
typedef __attribute__((ext_vector_type(8)))  float  v8f;

union FragCast { uint4 q[2]; v16bf bf; };

__device__ __forceinline__ unsigned short f32_to_bf16(float f) {
  union { float f; unsigned u; } x; x.f = f;
  unsigned r = x.u + 0x7FFFu + ((x.u >> 16) & 1u);   // round-to-nearest-even
  return (unsigned short)(r >> 16);
}

__device__ __forceinline__ float wave_sum(float v) {
  #pragma unroll
  for (int o = 16; o > 0; o >>= 1) v += __shfl_xor(v, o, 32);
  return v;
}
__device__ __forceinline__ float wave_max(float v) {
  #pragma unroll
  for (int o = 16; o > 0; o >>= 1) v = fmaxf(v, __shfl_xor(v, o, 32));
  return v;
}

// A fragment: 16x32 bf16, lane L(0..15) row M=L K=0..7|16..23, lane L+16 K=8..15|24..31
__device__ __forceinline__ v16bf load_frag_a(const unsigned short* __restrict__ base,
                                             int ld, int row0, int k0, int lane) {
  const int row  = row0 + (lane & 15);
  const int koff = (lane >> 4) << 3;                 // 0 or 8
  const unsigned short* p = base + (size_t)row * ld + k0 + koff;
  FragCast u;
  u.q[0] = *(const uint4*)(p);                       // K = koff .. koff+7
  u.q[1] = *(const uint4*)(p + 16);                  // K = koff+16 .. koff+23
  return u.bf;
}

// B fragment: 32x16 bf16, lane L(0..15) col N=L K=0..7|16..23, lane L+16 K=8..15|24..31.
// 'base' is indexed so that column n of B is contiguous in K at base[n*ld + k].
__device__ __forceinline__ v16bf load_frag_b(const unsigned short* __restrict__ base,
                                             int ld, int col0, int k0, int lane) {
  const int col  = col0 + (lane & 15);
  const int koff = (lane >> 4) << 3;                 // 0 or 8
  const unsigned short* p = base + (size_t)col * ld + k0 + koff;
  FragCast u;
  u.q[0] = *(const uint4*)(p);                       // K = koff .. koff+7
  u.q[1] = *(const uint4*)(p + 16);                  // K = koff+16 .. koff+23
  return u.bf;
}

__device__ __forceinline__ v8f wmma_bf16(v16bf a, v16bf b, v8f c) {
  return __builtin_amdgcn_wmma_f32_16x16x32_bf16(false, a, false, b, (short)0, c,
                                                 false, false);
}

// ---------------------------------------------------------------- converts
__global__ void cvt_bf16_kernel(const float* __restrict__ src,
                                unsigned short* __restrict__ dst, int n) {
  int i = blockIdx.x * blockDim.x + threadIdx.x;
  if (i < n) dst[i] = f32_to_bf16(src[i]);
}

// ------------------------------------------------- rank-1 score factors a,c
// a[b,s] = hs[b,s,:].U[:H] + U[H] ;  c[b,s] = hs[b,s,:].V[:H] + V[H]
__global__ void rowstats_kernel(const float* __restrict__ hs,
                                const float* __restrict__ U,
                                const float* __restrict__ V,
                                float* __restrict__ a, float* __restrict__ c) {
  const int wave = (blockIdx.x * blockDim.x + threadIdx.x) >> 5;  // 0..B*S-1
  const int lane = threadIdx.x & 31;
  const float* row = hs + (size_t)wave * HH;
  float sa = 0.f, sc = 0.f;
  for (int h = lane; h < HH; h += 32) {
    float x = row[h];
    sa = fmaf(x, U[h], sa);
    sc = fmaf(x, V[h], sc);
  }
  sa = wave_sum(sa);
  sc = wave_sum(sc);
  if (lane == 0) { a[wave] = sa + U[HH]; c[wave] = sc + V[HH]; }
}

// ------------------------------------------- fused softmax of rank-1 scores
// P[b,s,t] = softmax_t(a[b,s]*c[b,t]/8 + mask[b,t])  -> bf16
__global__ void softmax_probs_kernel(const float* __restrict__ a,
                                     const float* __restrict__ c,
                                     const float* __restrict__ mask,
                                     unsigned short* __restrict__ P) {
  const int wave = (blockIdx.x * blockDim.x + threadIdx.x) >> 5;  // 0..B*S-1
  const int lane = threadIdx.x & 31;
  const int b = wave >> 11;                                       // /S
  const float as = a[wave] * 0.125f;                              // 1/sqrt(64)
  const float* cb = c + (size_t)b * SS;
  const float* mb = mask + (size_t)b * SS;
  float mx = -3.4e38f;
  for (int t = lane; t < SS; t += 32) mx = fmaxf(mx, fmaf(as, cb[t], mb[t]));
  mx = wave_max(mx);
  float sum = 0.f;
  for (int t = lane; t < SS; t += 32) sum += __expf(fmaf(as, cb[t], mb[t]) - mx);
  sum = wave_sum(sum);
  const float inv = 1.f / sum;
  unsigned short* row = P + (size_t)wave * SS;
  for (int t = lane; t < SS; t += 32)
    row[t] = f32_to_bf16(__expf(fmaf(as, cb[t], mb[t]) - mx) * inv);
}

// --------------------------------------------------------------- GEMM 1
// VLt[b,h,t] = sum_k W[h,k]*hs[b,t,k] + bias[h]   (value layer, transposed)
__global__ __launch_bounds__(256) void gemm1_value_kernel(
    const unsigned short* __restrict__ Wbf,   // [H,H]
    const unsigned short* __restrict__ hsbf,  // [B,S,H]
    const float* __restrict__ bias,           // [H]
    unsigned short* __restrict__ vlt) {       // [B,H,S]
  const int wave = (blockIdx.x * blockDim.x + threadIdx.x) >> 5;
  const int lane = threadIdx.x & 31;
  const int b  = wave >> 10;                  // 32*32 wave-tiles per batch
  const int r  = wave & 1023;
  const int m0 = (r >> 5) << 5;               // h block, 32 rows
  const int n0 = (r & 31) << 6;               // t block, 64 cols
  const unsigned short* Bsrc = hsbf + (size_t)b * SS * HH;
  v8f acc[2][4] = {};
  for (int k0 = 0; k0 < HH; k0 += 32) {
    v16bf a0 = load_frag_a(Wbf, HH, m0,      k0, lane);
    v16bf a1 = load_frag_a(Wbf, HH, m0 + 16, k0, lane);
    v16bf b0 = load_frag_b(Bsrc, HH, n0,      k0, lane);
    v16bf b1 = load_frag_b(Bsrc, HH, n0 + 16, k0, lane);
    v16bf b2 = load_frag_b(Bsrc, HH, n0 + 32, k0, lane);
    v16bf b3 = load_frag_b(Bsrc, HH, n0 + 48, k0, lane);
    acc[0][0] = wmma_bf16(a0, b0, acc[0][0]);
    acc[0][1] = wmma_bf16(a0, b1, acc[0][1]);
    acc[0][2] = wmma_bf16(a0, b2, acc[0][2]);
    acc[0][3] = wmma_bf16(a0, b3, acc[0][3]);
    acc[1][0] = wmma_bf16(a1, b0, acc[1][0]);
    acc[1][1] = wmma_bf16(a1, b1, acc[1][1]);
    acc[1][2] = wmma_bf16(a1, b2, acc[1][2]);
    acc[1][3] = wmma_bf16(a1, b3, acc[1][3]);
  }
  unsigned short* C = vlt + (size_t)b * HH * SS;
  const int half8 = (lane >> 4) << 3;
  const int cn    = lane & 15;
  #pragma unroll
  for (int i = 0; i < 2; ++i)
    #pragma unroll
    for (int j = 0; j < 4; ++j)
      #pragma unroll
      for (int rr = 0; rr < 8; ++rr) {
        const int mm = m0 + 16 * i + rr + half8;
        const float v = acc[i][j][rr] + bias[mm];
        C[(size_t)mm * SS + (n0 + 16 * j + cn)] = f32_to_bf16(v);
      }
}

// --------------------------------------------------------------- GEMM 2
// out[b,s,h] = sum_t P[b,s,t] * VLt[b,h,t]
__global__ __launch_bounds__(256) void gemm2_context_kernel(
    const unsigned short* __restrict__ Pbf,   // [B,S,S]
    const unsigned short* __restrict__ vlt,   // [B,H,S]
    float* __restrict__ out) {                // [B,S,H]
  const int wave = (blockIdx.x * blockDim.x + threadIdx.x) >> 5;
  const int lane = threadIdx.x & 31;
  const int b  = wave >> 10;                  // 64*16 wave-tiles per batch
  const int r  = wave & 1023;
  const int m0 = (r >> 4) << 5;               // s block, 32 rows
  const int n0 = (r & 15) << 6;               // h block, 64 cols
  const unsigned short* A    = Pbf + (size_t)b * SS * SS;
  const unsigned short* Bsrc = vlt + (size_t)b * HH * SS;
  v8f acc[2][4] = {};
  for (int k0 = 0; k0 < SS; k0 += 32) {
    v16bf a0 = load_frag_a(A, SS, m0,      k0, lane);
    v16bf a1 = load_frag_a(A, SS, m0 + 16, k0, lane);
    v16bf b0 = load_frag_b(Bsrc, SS, n0,      k0, lane);
    v16bf b1 = load_frag_b(Bsrc, SS, n0 + 16, k0, lane);
    v16bf b2 = load_frag_b(Bsrc, SS, n0 + 32, k0, lane);
    v16bf b3 = load_frag_b(Bsrc, SS, n0 + 48, k0, lane);
    acc[0][0] = wmma_bf16(a0, b0, acc[0][0]);
    acc[0][1] = wmma_bf16(a0, b1, acc[0][1]);
    acc[0][2] = wmma_bf16(a0, b2, acc[0][2]);
    acc[0][3] = wmma_bf16(a0, b3, acc[0][3]);
    acc[1][0] = wmma_bf16(a1, b0, acc[1][0]);
    acc[1][1] = wmma_bf16(a1, b1, acc[1][1]);
    acc[1][2] = wmma_bf16(a1, b2, acc[1][2]);
    acc[1][3] = wmma_bf16(a1, b3, acc[1][3]);
  }
  float* C = out + (size_t)b * SS * HH;
  const int half8 = (lane >> 4) << 3;
  const int cn    = lane & 15;
  #pragma unroll
  for (int i = 0; i < 2; ++i)
    #pragma unroll
    for (int j = 0; j < 4; ++j)
      #pragma unroll
      for (int rr = 0; rr < 8; ++rr) {
        const int mm = m0 + 16 * i + rr + half8;
        C[(size_t)mm * HH + (n0 + 16 * j + cn)] = acc[i][j][rr];
      }
}

extern "C" void kernel_launch(void* const* d_in, const int* in_sizes, int n_in,
                              void* d_out, int out_size, void* d_ws, size_t ws_size,
                              hipStream_t stream) {
  const float* hs    = (const float*)d_in[0];   // [B,S,H]
  const float* mask  = (const float*)d_in[1];   // [B,S]
  const float* Wv    = (const float*)d_in[2];   // [H,H]
  const float* bias  = (const float*)d_in[3];   // [H]
  const float* U     = (const float*)d_in[4];   // [H+1,1]
  const float* V     = (const float*)d_in[5];   // [1,H+1]
  float* out = (float*)d_out;                   // [B,S,H]

  // workspace partition
  char* w = (char*)d_ws;
  unsigned short* hs_bf = (unsigned short*)w; w += (size_t)BB * SS * HH * 2;  // 16 MiB
  unsigned short* w_bf  = (unsigned short*)w; w += (size_t)HH * HH * 2;       //  2 MiB
  unsigned short* vlt   = (unsigned short*)w; w += (size_t)BB * HH * SS * 2;  // 16 MiB
  unsigned short* p_bf  = (unsigned short*)w; w += (size_t)BB * SS * SS * 2;  // 32 MiB
  float* a_v = (float*)w; w += (size_t)BB * SS * 4;
  float* c_v = (float*)w; w += (size_t)BB * SS * 4;

  const int n_hs = BB * SS * HH;
  const int n_w  = HH * HH;

  cvt_bf16_kernel<<<(n_hs + 255) / 256, 256, 0, stream>>>(hs, hs_bf, n_hs);
  cvt_bf16_kernel<<<(n_w + 255) / 256, 256, 0, stream>>>(Wv, w_bf, n_w);

  // B*S waves, 8 waves per 256-thread block
  rowstats_kernel<<<(BB * SS) / 8, 256, 0, stream>>>(hs, U, V, a_v, c_v);
  softmax_probs_kernel<<<(BB * SS) / 8, 256, 0, stream>>>(a_v, c_v, mask, p_bf);

  // 4096 waves each, 8 waves per block -> 512 blocks
  gemm1_value_kernel<<<512, 256, 0, stream>>>(w_bf, hs_bf, bias, vlt);
  gemm2_context_kernel<<<512, 256, 0, stream>>>(p_bf, vlt, out);
}
